// TransformerLayerShard_22643067585218
// MI455X (gfx1250) — matrix operand order repaired and checked by reading, not compiled, not run on hardware
//
#include <hip/hip_runtime.h>
#include <hip/hip_bf16.h>

typedef __attribute__((ext_vector_type(16))) _Float16 v16h;
typedef __attribute__((ext_vector_type(8)))  float    v8f;

#define Dmodel 4096
#define Hn 16
#define DHn 256
#define Rrot 64
#define Tn 2048

__device__ __forceinline__ float gelu_tanh(float x) {
  const float x3 = x * x * x;
  return 0.5f * x * (1.f + tanhf(0.7978845608028654f * (x + 0.044715f * x3)));
}

// 16-byte global -> LDS async copy (CDNA5, tracked by ASYNCcnt)
__device__ __forceinline__ void async_copy_b128(unsigned lds_off, const void* gptr) {
  asm volatile("global_load_async_to_lds_b128 %0, %1, off"
               :: "v"(lds_off), "v"((unsigned long long)gptr)
               : "memory");
}
__device__ __forceinline__ void wait_async0() {
  asm volatile("s_wait_asynccnt 0x0" ::: "memory");
}

// ---------------- LayerNorm (writes f16 xn) ----------------
__global__ __launch_bounds__(256)
void ln_kernel(const float* __restrict__ x, const float* __restrict__ gamma,
               const float* __restrict__ beta, _Float16* __restrict__ xnh) {
  __shared__ float s1[256];
  __shared__ float s2[256];
  const int t = blockIdx.x;
  const int tid = threadIdx.x;
  const float* row = x + (size_t)t * Dmodel;
  float a = 0.f, b = 0.f;
  for (int j = tid; j < Dmodel; j += 256) { float vv = row[j]; a += vv; b += vv * vv; }
  s1[tid] = a; s2[tid] = b;
  __syncthreads();
  for (int off = 128; off > 0; off >>= 1) {
    if (tid < off) { s1[tid] += s1[tid + off]; s2[tid] += s2[tid + off]; }
    __syncthreads();
  }
  const float mean = s1[0] * (1.f / Dmodel);
  const float var  = s2[0] * (1.f / Dmodel) - mean * mean;
  const float inv  = rsqrtf(var + 1e-5f);
  _Float16* out = xnh + (size_t)t * Dmodel;
  for (int j = tid; j < Dmodel; j += 256)
    out[j] = (_Float16)((row[j] - mean) * inv * gamma[j] + beta[j]);
}

// ---------------- f32 -> f16 weight conversion ----------------
__global__ __launch_bounds__(256)
void cvt_kernel(const float* __restrict__ src, _Float16* __restrict__ dst, int n4) {
  const int i = blockIdx.x * 256 + threadIdx.x;   // float4 index
  if (i < n4) {
    const float4 f = ((const float4*)src)[i];
    dst[4 * i + 0] = (_Float16)f.x;
    dst[4 * i + 1] = (_Float16)f.y;
    dst[4 * i + 2] = (_Float16)f.z;
    dst[4 * i + 3] = (_Float16)f.w;
  }
}

// ---------------- RoPE (in place on f16 q and k) ----------------
__global__ void rope_kernel(_Float16* __restrict__ q, _Float16* __restrict__ kk) {
  const int idx = blockIdx.x * blockDim.x + threadIdx.x;
  const int per = Tn * Hn * (Rrot / 2);
  if (idx >= 2 * per) return;
  _Float16* base = (idx < per) ? q : kk;
  int rem = (idx < per) ? idx : idx - per;
  const int i = rem & 31;       rem >>= 5;
  const int h = rem & (Hn - 1);
  const int t = rem >> 4;
  const float inv = __powf(10000.f, -((float)(2 * i)) / (float)Rrot);
  const float ang = (float)t * inv;
  const float c = __cosf(ang), s = __sinf(ang);
  _Float16* p = base + ((size_t)t * Hn + h) * DHn + 2 * i;
  const float xe = (float)p[0], xo = (float)p[1];
  p[0] = (_Float16)(xe * c - xo * s);
  p[1] = (_Float16)(xo * c + xe * s);
}

// ---------------- V transpose: vh[s][h*256+d] -> vT[h*256+d][s] ----------------
__global__ __launch_bounds__(256)
void vtrans_kernel(const _Float16* __restrict__ vh, _Float16* __restrict__ vT) {
  const int idx = blockIdx.x * 256 + threadIdx.x;   // over T*D
  const int d = idx & (Dmodel - 1);
  const int s = idx >> 12;
  vT[(size_t)d * Tn + s] = vh[idx];
}

// ------- WMMA GEMM on f16 operands, double-buffered LDS, async A staging -------
// C = A(f16,MxK) @ B(f16,KxN).
// MODE 0: f32 store; 1: bias+gelu -> f16; 2: bias+add -> f32; 3: f16 store
template <int MODE>
__global__ __launch_bounds__(256)
void gemm_kernel(const _Float16* __restrict__ A, const _Float16* __restrict__ B,
                 const float* __restrict__ bias, float* __restrict__ Cf,
                 _Float16* __restrict__ Ch, int N, int K) {
  __shared__ _Float16 As[2][128][40];   // A tile, rows padded to 80B (16B-aligned segs)
  __shared__ _Float16 Bs[2][64][40];    // B tile TRANSPOSED: Bs[buf][n][k]
  const int tid  = threadIdx.x;
  const int lane = tid & 31;
  const int wid  = tid >> 5;
  const int wm   = wid & 3;             // 4 m-waves x 32 rows
  const int wn   = wid >> 2;            // 2 n-waves x 32 cols
  const int m_block = blockIdx.y * 128;
  const int n_block = blockIdx.x * 64;

  v8f acc[2][2] = {};
  const int nl    = lane & 15;
  const int abase = (lane < 16) ? 0 : 16;  // A-fragment K start
  const int bbase = (lane < 16) ? 0 : 16;  // B-fragment K start

  auto stageA = [&](int buf, int kt) {     // 128x32 f16 = 512 x 16B async chunks
    #pragma unroll
    for (int u = 0; u < 2; ++u) {
      const int c   = tid * 2 + u;
      const int row = c >> 2;
      const int seg = c & 3;
      const unsigned loff = (unsigned)(unsigned long long)&As[buf][row][seg * 8];
      async_copy_b128(loff, A + (size_t)(m_block + row) * K + kt + seg * 8);
    }
  };
  auto stageB = [&](int buf, int kt) {     // 32x64 f16, b128 load + transposed scatter
    const int kr = tid >> 3;               // k row 0..31
    const int c0 = (tid & 7) * 8;          // 8 contiguous cols
    const _Float16* src = B + (size_t)(kt + kr) * N + n_block + c0;
    #pragma unroll
    for (int j = 0; j < 8; ++j) Bs[buf][c0 + j][kr] = src[j];
  };

  stageA(0, 0);
  stageB(0, 0);
  int cur = 0;

  for (int kt = 0; kt < K; kt += 32) {
    wait_async0();
    __syncthreads();                       // cur buffer ready for everyone
    const int nxt = cur ^ 1;
    if (kt + 32 < K) { stageA(nxt, kt + 32); stageB(nxt, kt + 32); }

    v16h af[2];
    #pragma unroll
    for (int mi = 0; mi < 2; ++mi) {
      const int row = wm * 32 + mi * 16 + nl;
      #pragma unroll
      for (int j = 0; j < 8; ++j) {
        af[mi][j]     = As[cur][row][abase + j];
        af[mi][8 + j] = As[cur][row][abase + 8 + j];
      }
    }
    #pragma unroll
    for (int ni = 0; ni < 2; ++ni) {
      v16h bf;
      const int col = wn * 32 + ni * 16 + nl;
      #pragma unroll
      for (int j = 0; j < 16; ++j) bf[j] = Bs[cur][col][bbase + j];
      #pragma unroll
      for (int mi = 0; mi < 2; ++mi)
        acc[mi][ni] = __builtin_amdgcn_wmma_f32_16x16x32_f16(
            false, af[mi], false, bf, (short)0, acc[mi][ni], false, false);
    }
    cur = nxt;
  }

  const int roff = (lane < 16) ? 0 : 8;
  #pragma unroll
  for (int mi = 0; mi < 2; ++mi) {
    #pragma unroll
    for (int ni = 0; ni < 2; ++ni) {
      const int col = n_block + wn * 32 + ni * 16 + nl;
      #pragma unroll
      for (int r = 0; r < 8; ++r) {
        const int row = m_block + wm * 32 + mi * 16 + r + roff;
        const float val = acc[mi][ni][r];
        if constexpr (MODE == 1) {
          Ch[(size_t)row * N + col] = (_Float16)gelu_tanh(val + bias[col]);
        } else if constexpr (MODE == 2) {
          Cf[(size_t)row * N + col] += val + bias[col];
        } else if constexpr (MODE == 3) {
          Ch[(size_t)row * N + col] = (_Float16)val;
        } else {
          Cf[(size_t)row * N + col] = val;
        }
      }
    }
  }
}

// ---------------- Flash attention: 1 wave per (head, 16-query tile) ----------------
// q,k: f16 [T][H][DH]; vT: f16 [H][DH][T]; ctxh: f16 [T][H][DH]
__global__ __launch_bounds__(32)
void attn_kernel(const _Float16* __restrict__ q, const _Float16* __restrict__ k,
                 const _Float16* __restrict__ vT, _Float16* __restrict__ ctxh) {
  __shared__ _Float16 Ps[16][32];
  const int h    = blockIdx.y;
  const int qb   = blockIdx.x * 16;
  const int lane = threadIdx.x;
  const int nl    = lane & 15;
  const int abase = (lane < 16) ? 0 : 16;
  const int bbase = (lane < 16) ? 0 : 16;
  const int roff  = (lane < 16) ? 0 : 8;

  v16h qf[8];
  {
    const _Float16* qrow = q + ((size_t)(qb + nl) * Hn + h) * DHn;
    #pragma unroll
    for (int kk = 0; kk < 8; ++kk)
      #pragma unroll
      for (int j = 0; j < 8; ++j) {
        qf[kk][j]     = qrow[kk * 32 + abase + j];
        qf[kk][8 + j] = qrow[kk * 32 + abase + 8 + j];
      }
  }

  v8f oacc[16] = {};
  float rmax[8], rsum[8];
  #pragma unroll
  for (int r = 0; r < 8; ++r) { rmax[r] = -1e30f; rsum[r] = 0.f; }

  for (int kc = 0; kc <= qb + 15; kc += 32) {
    v8f s0 = {}, s1 = {};
    #pragma unroll
    for (int kk = 0; kk < 8; ++kk) {
      const _Float16* k0 = k + ((size_t)(kc + nl)      * Hn + h) * DHn + kk * 32 + bbase;
      const _Float16* k1 = k + ((size_t)(kc + 16 + nl) * Hn + h) * DHn + kk * 32 + bbase;
      v16h bf0, bf1;
      #pragma unroll
      for (int j = 0; j < 16; ++j) { bf0[j] = k0[j]; bf1[j] = k1[j]; }
      s0 = __builtin_amdgcn_wmma_f32_16x16x32_f16(false, qf[kk], false, bf0, (short)0, s0, false, false);
      s1 = __builtin_amdgcn_wmma_f32_16x16x32_f16(false, qf[kk], false, bf1, (short)0, s1, false, false);
    }

    const float scale = 0.0625f;     // 1/sqrt(256)
    #pragma unroll
    for (int r = 0; r < 8; ++r) {
      const int trow = qb + r + roff;
      float a = s0[r] * scale + ((kc + nl)      > trow ? -1e10f : 0.f);
      float b = s1[r] * scale + ((kc + 16 + nl) > trow ? -1e10f : 0.f);
      float mx = fmaxf(a, b);
      for (int off = 1; off < 16; off <<= 1) mx = fmaxf(mx, __shfl_xor(mx, off, 32));
      const float nm   = fmaxf(rmax[r], mx);
      const float corr = __expf(rmax[r] - nm);
      const float p0 = __expf(a - nm);
      const float p1 = __expf(b - nm);
      float ps = p0 + p1;
      for (int off = 1; off < 16; off <<= 1) ps += __shfl_xor(ps, off, 32);
      rsum[r] = rsum[r] * corr + ps;
      rmax[r] = nm;
      #pragma unroll
      for (int dt = 0; dt < 16; ++dt) oacc[dt][r] *= corr;
      Ps[r + roff][nl]      = (_Float16)p0;
      Ps[r + roff][16 + nl] = (_Float16)p1;
    }
    __syncthreads();
    v16h pf;
    #pragma unroll
    for (int j = 0; j < 8; ++j) {
      pf[j]     = Ps[nl][abase + j];
      pf[8 + j] = Ps[nl][abase + 8 + j];
    }
    __syncthreads();
    #pragma unroll
    for (int dt = 0; dt < 16; ++dt) {
      // V fragment: lane = d column (dt*16+nl), contiguous along keys in vT
      const _Float16* vt = vT + ((size_t)h * DHn + dt * 16 + nl) * Tn + kc + bbase;
      v16h vf;
      #pragma unroll
      for (int j = 0; j < 16; ++j) vf[j] = vt[j];
      oacc[dt] = __builtin_amdgcn_wmma_f32_16x16x32_f16(false, pf, false, vf, (short)0, oacc[dt], false, false);
    }
  }

  #pragma unroll
  for (int dt = 0; dt < 16; ++dt)
    #pragma unroll
    for (int r = 0; r < 8; ++r) {
      const int trow = qb + r + roff;
      ctxh[((size_t)trow * Hn + h) * DHn + dt * 16 + nl] = (_Float16)(oacc[dt][r] / rsum[r]);
    }
}

extern "C" void kernel_launch(void* const* d_in, const int* in_sizes, int n_in,
                              void* d_out, int out_size, void* d_ws, size_t ws_size,
                              hipStream_t stream) {
  const float* x         = (const float*)d_in[0];
  const float* ln_scale  = (const float*)d_in[1];
  const float* ln_offset = (const float*)d_in[2];
  const float* wq = (const float*)d_in[3];
  const float* wk = (const float*)d_in[4];
  const float* wv = (const float*)d_in[5];
  const float* wo = (const float*)d_in[6];
  const float* w1 = (const float*)d_in[7];
  const float* b1 = (const float*)d_in[8];
  const float* w2 = (const float*)d_in[9];
  const float* b2 = (const float*)d_in[10];
  float* out = (float*)d_out;

  const size_t TD  = (size_t)Tn * Dmodel;           // 8,388,608
  const size_t WFF = (size_t)Dmodel * 4 * Dmodel;   // 67,108,864 (largest weight)

  char* w = (char*)d_ws;
  _Float16* wbuf  = (_Float16*)w;  w += WFF * 2;    // 128MB shared f16 weight staging
  _Float16* xnh   = (_Float16*)w;  w += TD * 2;     // 16MB
  _Float16* ctxh  = (_Float16*)w;  w += TD * 2;     // 16MB
  _Float16* hbufh = (_Float16*)w;  w += TD * 4 * 2; // 64MB
  _Float16* qh    = (_Float16*)w;  w += TD * 2;     // 16MB
  _Float16* kh    = (_Float16*)w;  w += TD * 2;     // 16MB
  _Float16* vh    = (_Float16*)w;  w += TD * 2;     // 16MB
  _Float16* vTh   = (_Float16*)w;  /* 16MB */

  const int nDD4  = (Dmodel * Dmodel) / 4;          // float4 count for DxD
  const int nFF4  = (int)(WFF / 4);                 // float4 count for D x 4D
  dim3 blk(256);
  dim3 g1(Dmodel / 64, Tn / 128);
  dim3 g2((4 * Dmodel) / 64, Tn / 128);

  ln_kernel<<<Tn, 256, 0, stream>>>(x, ln_scale, ln_offset, xnh);

  cvt_kernel<<<nDD4 / 256, blk, 0, stream>>>(wq, wbuf, nDD4);
  gemm_kernel<3><<<g1, blk, 0, stream>>>(xnh, wbuf, nullptr, nullptr, qh, Dmodel, Dmodel);
  cvt_kernel<<<nDD4 / 256, blk, 0, stream>>>(wk, wbuf, nDD4);
  gemm_kernel<3><<<g1, blk, 0, stream>>>(xnh, wbuf, nullptr, nullptr, kh, Dmodel, Dmodel);
  cvt_kernel<<<nDD4 / 256, blk, 0, stream>>>(wv, wbuf, nDD4);
  gemm_kernel<3><<<g1, blk, 0, stream>>>(xnh, wbuf, nullptr, nullptr, vh, Dmodel, Dmodel);

  const int rope_threads = 2 * Tn * Hn * (Rrot / 2);
  rope_kernel<<<(rope_threads + 255) / 256, 256, 0, stream>>>(qh, kh);
  vtrans_kernel<<<(int)(TD / 256), blk, 0, stream>>>(vh, vTh);

  attn_kernel<<<dim3(Tn / 16, Hn), dim3(32), 0, stream>>>(qh, kh, vTh, ctxh);

  cvt_kernel<<<nDD4 / 256, blk, 0, stream>>>(wo, wbuf, nDD4);
  gemm_kernel<0><<<g1, blk, 0, stream>>>(ctxh, wbuf, nullptr, out, nullptr, Dmodel, Dmodel);

  cvt_kernel<<<nFF4 / 256, blk, 0, stream>>>(w1, wbuf, nFF4);
  gemm_kernel<1><<<g2, blk, 0, stream>>>(xnh, wbuf, b1, nullptr, hbufh, 4 * Dmodel, Dmodel);
  cvt_kernel<<<nFF4 / 256, blk, 0, stream>>>(w2, wbuf, nFF4);
  gemm_kernel<2><<<g1, blk, 0, stream>>>(hbufh, wbuf, b2, out, nullptr, Dmodel, 4 * Dmodel);

  (void)in_sizes; (void)n_in; (void)out_size; (void)ws_size;
}
